// DeepQTH_34437047779388
// MI455X (gfx1250) — compile-verified, matrix operand order checked by reading, not run on hardware
//
#include <hip/hip_runtime.h>
#include <hip/hip_bf16.h>

// ---------------------------------------------------------------------------
// Layout constants
// ---------------------------------------------------------------------------
#define K1_ROWS 128
#define ZK_PAD  264   // z / Wf / Ws rows: 256 K + 8 pad bf16 (528 B, 16B aligned)
#define K2_EDGES 64
#define HK_PAD   264  // h / W1 rows: K 240 -> 256 (+8 stride pad)
#define H2K_PAD  136  // hid / W2 rows: K 128 (+8 stride pad), 272 B

// ---------------------------------------------------------------------------
// Types for CDNA5 WMMA (wave32): D(16x16 f32) = A(16x32 bf16) * B(32x16 bf16) + C
// ---------------------------------------------------------------------------
typedef __bf16 bf16_t;
typedef __attribute__((ext_vector_type(16))) __bf16 v16bf;
typedef __attribute__((ext_vector_type(8)))  float  v8f;
typedef __attribute__((ext_vector_type(4)))  unsigned v4u;
typedef __attribute__((ext_vector_type(4)))  float  v4f;
typedef __attribute__((ext_vector_type(4)))  int    v4i;

union FragU { v4u q[2]; v16bf v; };

// ---------------------------------------------------------------------------
// Async global->LDS copy path (CDNA5 GLOBAL_LOAD_ASYNC_TO_LDS_B128, ASYNCcnt)
// Builtin signature (from clang diagnostic): (v4i addrspace(1)*, v4i addrspace(3)*, int, int)
// ---------------------------------------------------------------------------
#if defined(__has_builtin)
#if __has_builtin(__builtin_amdgcn_global_load_async_to_lds_b128)
#define HAVE_ASYNC_LDS 1
#endif
#endif

typedef __attribute__((address_space(1))) v4i gv4i_t;   // global int4
typedef __attribute__((address_space(3))) v4i lv4i_t;   // LDS int4

__device__ __forceinline__ void copy16_to_lds(bf16_t* dst, const bf16_t* src) {
#ifdef HAVE_ASYNC_LDS
    __builtin_amdgcn_global_load_async_to_lds_b128(
        (gv4i_t*)src, (lv4i_t*)dst, /*offset=*/0, /*cpol=*/0);
#else
    *(v4u*)dst = *(const v4u*)src;
#endif
}

__device__ __forceinline__ void wait_async_copies() {
#ifdef HAVE_ASYNC_LDS
#if __has_builtin(__builtin_amdgcn_s_wait_asynccnt)
    __builtin_amdgcn_s_wait_asynccnt(0);
#else
    asm volatile("s_wait_asynccnt 0x0" ::: "memory");
#endif
#endif
}

__device__ __forceinline__ bf16_t f2b(float f) {
    unsigned u = __float_as_uint(f);
    u = (u + 0x7FFFu + ((u >> 16) & 1u)) >> 16;           // round-to-nearest-even
    unsigned short s = (unsigned short)u;
    return __builtin_bit_cast(bf16_t, s);
}

// pack two fp32 -> packed bf16 pair (lo in [15:0], hi in [31:16]), RNE
__device__ __forceinline__ unsigned pack2(float lo, float hi) {
    unsigned a = __float_as_uint(lo), b = __float_as_uint(hi);
    a = (a + 0x7FFFu + ((a >> 16) & 1u)) >> 16;
    b = (b + 0x7FFFu + ((b >> 16) & 1u)) & 0xFFFF0000u;
    return a | b;
}

// convert 8 consecutive fp32 (two float4) -> one uint4 of 8 bf16
__device__ __forceinline__ v4u cvt8(v4f a, v4f b) {
    v4u r;
    r.x = pack2(a.x, a.y); r.y = pack2(a.z, a.w);
    r.z = pack2(b.x, b.y); r.w = pack2(b.z, b.w);
    return r;
}

// load 8 fp32 from global (16B aligned), convert, store 8 bf16 to LDS (16B aligned)
__device__ __forceinline__ void gcopy8(bf16_t* dst, const float* src) {
    v4f a = *(const v4f*)src;
    v4f b = *(const v4f*)(src + 4);
    *(v4u*)dst = cvt8(a, b);
}

__device__ __forceinline__ void zero8(bf16_t* dst) {
    v4u z = {0u, 0u, 0u, 0u};
    *(v4u*)dst = z;
}

// Load one 16x32 bf16 fragment (A or B) from an LDS row laid out [row][K].
// Per ISA 7.12.2: lanes 0-15 take K {0..7, 16..23}, lanes 16-31 take K {8..15, 24..31}
// => bytes {0..15, 32..47} (+16 for hi half). Two ds_load_b128 per fragment.
__device__ __forceinline__ v16bf lds_frag(const bf16_t* rowBase, int kstep, int hi) {
    const char* p = (const char*)rowBase + kstep * 64 + (hi << 4);
    FragU f;
    f.q[0] = *(const v4u*)(p);
    f.q[1] = *(const v4u*)(p + 32);
    return f.v;
}

__device__ __forceinline__ v8f wmma_bf16(v16bf a, v16bf b, v8f c) {
    return __builtin_amdgcn_wmma_f32_16x16x32_bf16(
        /*neg_a=*/false, a, /*neg_b=*/false, b,
        /*c_mod=*/(short)0, c, /*reuse_a=*/false, /*reuse_b=*/false);
}

__device__ __forceinline__ v8f v8f_zero() {
    v8f z = {0.f, 0.f, 0.f, 0.f, 0.f, 0.f, 0.f, 0.f};
    return z;
}

__device__ __forceinline__ float fsigmoid(float x) {
    return 1.0f / (1.0f + __expf(-x));
}
__device__ __forceinline__ float fsoftplus(float x) {
    float ax = fabsf(x);
    return fmaxf(x, 0.0f) + __logf(1.0f + __expf(-ax));
}

// native no-return fp32 global atomic add (STOREcnt-tracked; drained at s_endpgm)
__device__ __forceinline__ void global_atomic_fadd(float* p, float v) {
    asm volatile("global_atomic_add_f32 %0, %1, off" : : "v"(p), "v"(v) : "memory");
}

// ---------------------------------------------------------------------------
// Kernel 0: zero the agg accumulator (2*NE x 64 floats) in workspace
// ---------------------------------------------------------------------------
__global__ void k0_zero(float* __restrict__ p, long n4) {
    long i = (long)blockIdx.x * blockDim.x + threadIdx.x;
    long stride = (long)gridDim.x * blockDim.x;
    v4f z = {0.f, 0.f, 0.f, 0.f};
    v4f* p4 = (v4f*)p;
    for (; i < n4; i += stride) p4[i] = z;
}

// ---------------------------------------------------------------------------
// Kernel P: one-time weight prep -> bf16, transposed [N][K_pad], pad zeroed,
// stored in workspace so worker blocks fill LDS with raw b128 copies.
// ---------------------------------------------------------------------------
__global__ void k_prep_weights(const float* __restrict__ Wf, const float* __restrict__ Ws,
                               const float* __restrict__ W1, const float* __restrict__ W2,
                               bf16_t* __restrict__ wfT, bf16_t* __restrict__ wsT,
                               bf16_t* __restrict__ w1T, bf16_t* __restrict__ w2T) {
    int tid = blockIdx.x * blockDim.x + threadIdx.x;
    int stride = gridDim.x * blockDim.x;
    for (int i = tid; i < 64 * ZK_PAD; i += stride) {
        int n = i / ZK_PAD, k = i % ZK_PAD;
        bool ok = k < 256;
        wfT[i] = ok ? f2b(Wf[k * 64 + n]) : f2b(0.f);
        wsT[i] = ok ? f2b(Ws[k * 64 + n]) : f2b(0.f);
    }
    for (int i = tid; i < 128 * HK_PAD; i += stride) {
        int n = i / HK_PAD, k = i % HK_PAD;
        w1T[i] = (k < 240) ? f2b(W1[k * 128 + n]) : f2b(0.f);
    }
    for (int i = tid; i < 144 * H2K_PAD; i += stride) {
        int n = i / H2K_PAD, k = i % H2K_PAD;
        w2T[i] = (k < 128) ? f2b(W2[k * 144 + n]) : f2b(0.f);
    }
}

// ---------------------------------------------------------------------------
// Kernel 1: gather z[128,256] -> gated MLP (two bf16 WMMA GEMMs, K=256, N=64)
//           -> sigmoid*softplus*decay -> native atomic scatter into agg
// 256 threads = 8 waves, 16 rows per wave.
// ---------------------------------------------------------------------------
__global__ __launch_bounds__(256) void k1_gate_scatter(
    const float* __restrict__ atom_fea, const float* __restrict__ edge_fea,
    const int* __restrict__ sub_atom_idx, const int* __restrict__ sub_edge_idx,
    const float* __restrict__ sub_edge_ang, const int* __restrict__ sub_index,
    const float* __restrict__ distance,
    const bf16_t* __restrict__ wfT, const bf16_t* __restrict__ wsT,
    const float* __restrict__ bfv, const float* __restrict__ bsv,
    float* __restrict__ agg, int NS_)
{
    __shared__ bf16_t sWf[64 * ZK_PAD];      // Wf^T [n][k] bf16
    __shared__ bf16_t sWs[64 * ZK_PAD];      // Ws^T [n][k] bf16
    __shared__ bf16_t sZ[K1_ROWS * ZK_PAD];  // z tile [m][k] bf16
    __shared__ float  sDecay[K1_ROWS];
    __shared__ int    sDst[K1_ROWS];
    __shared__ float  sBf[64], sBs[64];

    const int tid = threadIdx.x;

    // ---- async b128 copy of prepped weight images into LDS (ASYNCcnt) ----
    // Issued first so the copies overlap the z-tile gather below.
    {
        #pragma unroll
        for (int j = 0; j < (64 * ZK_PAD) / 8 / 256; ++j) {  // 2112/256 -> 8 full
            int i = j * 256 + tid;
            copy16_to_lds(sWf + i * 8, wfT + i * 8);
            copy16_to_lds(sWs + i * 8, wsT + i * 8);
        }
        int i = 8 * 256 + tid;                               // tail 64
        if (i < (64 * ZK_PAD) / 8) {
            copy16_to_lds(sWf + i * 8, wfT + i * 8);
            copy16_to_lds(sWs + i * 8, wsT + i * 8);
        }
    }
    if (tid < 64) { sBf[tid] = bfv[tid]; sBs[tid] = bsv[tid]; }

    // ---- gather z tile: row r = tid>>1, 2 threads per row (128 cols each) ----
    {
        int r = tid >> 1;
        int half = tid & 1;
        long s = (long)blockIdx.x * K1_ROWS + r;
        bool valid = s < (long)NS_;
        bf16_t* zr = &sZ[r * ZK_PAD];
        if (valid) {
            int i0 = sub_atom_idx[2 * s];
            int i1 = sub_atom_idx[2 * s + 1];
            int ei = sub_edge_idx[s];
            if (half == 0) {
                const float* a0 = atom_fea + (long)i0 * 64;
                const float* a1 = atom_fea + (long)i1 * 64;
                #pragma unroll
                for (int g = 0; g < 8; ++g) gcopy8(zr + g * 8, a0 + g * 8);
                #pragma unroll
                for (int g = 0; g < 8; ++g) gcopy8(zr + 64 + g * 8, a1 + g * 8);
                float d = distance[ei];
                sDecay[r] = __expf(-(d * d) * (1.0f / 18.0f));
                sDst[r]   = sub_index[s];
            } else {
                const float* ef = edge_fea + (long)ei * 112;
                const float* an = sub_edge_ang + s * 16;
                #pragma unroll
                for (int g = 0; g < 14; ++g) gcopy8(zr + 128 + g * 8, ef + g * 8);
                gcopy8(zr + 240, an);
                gcopy8(zr + 248, an + 8);
            }
        } else {
            #pragma unroll
            for (int g = 0; g < 16; ++g) zero8(zr + half * 128 + g * 8);
            if (half == 0) { sDecay[r] = 0.0f; sDst[r] = 0; }
        }
    }
    wait_async_copies();      // drain ASYNCcnt before the barrier (barriers don't)
    __syncthreads();

    // ---- WMMA: 16 rows/wave, N=64 (4 tiles), K=256 (8 steps) for F and S ----
    const int lane = tid & 31;
    const int wave = tid >> 5;
    const int hi   = lane >> 4;
    const int ln   = lane & 15;

    const bf16_t* aRow = &sZ[(wave * 16 + ln) * ZK_PAD];
    v8f accF[4], accS[4];
    #pragma unroll
    for (int nt = 0; nt < 4; ++nt) { accF[nt] = v8f_zero(); accS[nt] = v8f_zero(); }

    #pragma unroll
    for (int ks = 0; ks < 8; ++ks) {
        v16bf a = lds_frag(aRow, ks, hi);
        v16bf bF[4], bS[4];
        #pragma unroll
        for (int nt = 0; nt < 4; ++nt) {
            bF[nt] = lds_frag(&sWf[(nt * 16 + ln) * ZK_PAD], ks, hi);
            bS[nt] = lds_frag(&sWs[(nt * 16 + ln) * ZK_PAD], ks, hi);
        }
        #pragma unroll
        for (int nt = 0; nt < 4; ++nt) {
            accF[nt] = wmma_bf16(a, bF[nt], accF[nt]);
            accS[nt] = wmma_bf16(a, bS[nt], accS[nt]);
        }
    }

    // ---- epilogue: gate, decay, native fp32 atomic scatter-add ----
    #pragma unroll
    for (int nt = 0; nt < 4; ++nt) {
        int ncol = nt * 16 + ln;
        float bF = sBf[ncol], bS = sBs[ncol];
        #pragma unroll
        for (int v = 0; v < 8; ++v) {
            int m = v + hi * 8;             // C/D layout: VGPR v <-> row v (+8 hi)
            int r = wave * 16 + m;
            float g = fsigmoid(accF[nt][v] + bF) * fsoftplus(accS[nt][v] + bS)
                      * sDecay[r];
            global_atomic_fadd(agg + (long)sDst[r] * 64 + ncol, g);
        }
    }
}

// ---------------------------------------------------------------------------
// Kernel 2: edge MLP  out = silu(h @ W1 + b1) @ W2 + b2
// h[e,256pad] = [agg[2e] | agg[2e+1] | edge_fea[e] | 0]; 64 edges/block, 4 waves.
// ---------------------------------------------------------------------------
__global__ __launch_bounds__(128) void k2_edge_mlp(
    const float* __restrict__ edge_fea, const float* __restrict__ agg,
    const bf16_t* __restrict__ w1T, const bf16_t* __restrict__ w2T,
    const float* __restrict__ b1, const float* __restrict__ b2,
    float* __restrict__ out, int NE_)
{
    __shared__ bf16_t sW1[128 * HK_PAD];        // W1^T [n][k]
    __shared__ bf16_t sW2[144 * H2K_PAD];       // W2^T [n][k]
    __shared__ bf16_t sH[K2_EDGES * HK_PAD];    // h   [m][k]
    __shared__ bf16_t sHid[K2_EDGES * H2K_PAD]; // hid [m][k]
    __shared__ float  sB1[128], sB2[144];

    const int tid = threadIdx.x;   // 0..127

    // ---- async b128 copies of prepped weight images (overlap the gather) ----
    {
        #pragma unroll
        for (int j = 0; j < (128 * HK_PAD) / 8 / 128; ++j) {  // 4224/128 = 33
            int i = j * 128 + tid;
            copy16_to_lds(sW1 + i * 8, w1T + i * 8);
        }
        for (int i = tid; i < (144 * H2K_PAD) / 8; i += 128)  // 2448
            copy16_to_lds(sW2 + i * 8, w2T + i * 8);
    }
    sB1[tid] = b1[tid];
    sB2[tid] = b2[tid];
    if (tid < 16) sB2[128 + tid] = b2[128 + tid];

    // ---- gather h tile ----
    {
        int r = tid >> 1, half = tid & 1;
        long e = (long)blockIdx.x * K2_EDGES + r;
        bool valid = e < (long)NE_;
        bf16_t* hr = &sH[r * HK_PAD];
        if (half == 0) {
            if (valid) {
                const float* g0 = agg + (2 * e) * 64;
                const float* g1 = agg + (2 * e + 1) * 64;
                #pragma unroll
                for (int g = 0; g < 8; ++g) gcopy8(hr + g * 8, g0 + g * 8);
                #pragma unroll
                for (int g = 0; g < 8; ++g) gcopy8(hr + 64 + g * 8, g1 + g * 8);
            } else {
                #pragma unroll
                for (int g = 0; g < 16; ++g) zero8(hr + g * 8);
            }
        } else {
            if (valid) {
                const float* ef = edge_fea + e * 112;
                #pragma unroll
                for (int g = 0; g < 14; ++g) gcopy8(hr + 128 + g * 8, ef + g * 8);
            } else {
                #pragma unroll
                for (int g = 0; g < 14; ++g) zero8(hr + 128 + g * 8);
            }
            zero8(hr + 240);
            zero8(hr + 248);
        }
    }
    wait_async_copies();
    __syncthreads();

    const int lane = tid & 31;
    const int wave = tid >> 5;
    const int hi   = lane >> 4;
    const int ln   = lane & 15;

    // ---- stage 1: hid = silu(h @ W1 + b1);  K=256 (8 steps), N=128 (8 tiles) ----
    {
        const bf16_t* aRow = &sH[(wave * 16 + ln) * HK_PAD];
        v8f acc[8];
        #pragma unroll
        for (int nt = 0; nt < 8; ++nt) acc[nt] = v8f_zero();
        #pragma unroll
        for (int ks = 0; ks < 8; ++ks) {
            v16bf a = lds_frag(aRow, ks, hi);
            v16bf b[8];
            #pragma unroll
            for (int nt = 0; nt < 8; ++nt)
                b[nt] = lds_frag(&sW1[(nt * 16 + ln) * HK_PAD], ks, hi);
            #pragma unroll
            for (int nt = 0; nt < 8; ++nt)
                acc[nt] = wmma_bf16(a, b[nt], acc[nt]);
        }
        #pragma unroll
        for (int nt = 0; nt < 8; ++nt) {
            int ncol = nt * 16 + ln;
            float bb = sB1[ncol];
            #pragma unroll
            for (int v = 0; v < 8; ++v) {
                int m = v + hi * 8;
                float x = acc[nt][v] + bb;
                sHid[(wave * 16 + m) * H2K_PAD + ncol] = f2b(x * fsigmoid(x));
            }
        }
    }
    __syncthreads();

    // ---- stage 2: out = hid @ W2 + b2;  K=128 (4 steps), N=144 (9 tiles) ----
    {
        const bf16_t* aRow = &sHid[(wave * 16 + ln) * H2K_PAD];
        v8f acc[9];
        #pragma unroll
        for (int nt = 0; nt < 9; ++nt) acc[nt] = v8f_zero();
        #pragma unroll
        for (int ks = 0; ks < 4; ++ks) {
            v16bf a = lds_frag(aRow, ks, hi);
            v16bf b[9];
            #pragma unroll
            for (int nt = 0; nt < 9; ++nt)
                b[nt] = lds_frag(&sW2[(nt * 16 + ln) * H2K_PAD], ks, hi);
            #pragma unroll
            for (int nt = 0; nt < 9; ++nt)
                acc[nt] = wmma_bf16(a, b[nt], acc[nt]);
        }
        long ebase = (long)blockIdx.x * K2_EDGES + wave * 16;
        #pragma unroll
        for (int nt = 0; nt < 9; ++nt) {
            int ncol = nt * 16 + ln;
            float bb = sB2[ncol];
            #pragma unroll
            for (int v = 0; v < 8; ++v) {
                long e = ebase + v + hi * 8;
                if (e < (long)NE_) out[e * 144 + ncol] = acc[nt][v] + bb;
            }
        }
    }
}

// ---------------------------------------------------------------------------
// Launch
// ---------------------------------------------------------------------------
extern "C" void kernel_launch(void* const* d_in, const int* in_sizes, int n_in,
                              void* d_out, int out_size, void* d_ws, size_t ws_size,
                              hipStream_t stream) {
    const float* atom_fea     = (const float*)d_in[0];
    const float* edge_fea     = (const float*)d_in[1];
    const int*   sub_atom_idx = (const int*)  d_in[2];
    const int*   sub_edge_idx = (const int*)  d_in[3];
    const float* sub_edge_ang = (const float*)d_in[4];
    const int*   sub_index    = (const int*)  d_in[5];
    const float* distance     = (const float*)d_in[6];
    const float* Wf           = (const float*)d_in[7];
    const float* bfv          = (const float*)d_in[8];
    const float* Ws           = (const float*)d_in[9];
    const float* bsv          = (const float*)d_in[10];
    const float* W1           = (const float*)d_in[11];
    const float* b1           = (const float*)d_in[12];
    const float* W2           = (const float*)d_in[13];
    const float* b2           = (const float*)d_in[14];
    float*       out          = (float*)d_out;

    const int NS_ = in_sizes[3];            // sub_edge_idx: (NS,)
    const int NE_ = in_sizes[6];            // distance: (NE,)

    // ---- workspace layout ----
    char* wsB = (char*)d_ws;
    float* agg = (float*)wsB;                               // 2*NE x 64 fp32
    size_t off = ((size_t)2 * NE_ * 64 * sizeof(float) + 255) & ~(size_t)255;
    bf16_t* wfT = (bf16_t*)(wsB + off); off += (size_t)64  * ZK_PAD  * 2;
    bf16_t* wsT = (bf16_t*)(wsB + off); off += (size_t)64  * ZK_PAD  * 2;
    bf16_t* w1T = (bf16_t*)(wsB + off); off += (size_t)128 * HK_PAD  * 2;
    bf16_t* w2T = (bf16_t*)(wsB + off);

    long n4 = ((long)2 * NE_ * 64) / 4;
    k0_zero<<<2048, 256, 0, stream>>>(agg, n4);

    k_prep_weights<<<64, 256, 0, stream>>>(Wf, Ws, W1, W2, wfT, wsT, w1T, w2T);

    int blocks1 = (NS_ + K1_ROWS - 1) / K1_ROWS;
    k1_gate_scatter<<<blocks1, 256, 0, stream>>>(
        atom_fea, edge_fea, sub_atom_idx, sub_edge_idx, sub_edge_ang,
        sub_index, distance, wfT, wsT, bfv, bsv, agg, NS_);

    int blocks2 = (NE_ + K2_EDGES - 1) / K2_EDGES;
    k2_edge_mlp<<<blocks2, 128, 0, stream>>>(
        edge_fea, agg, w1T, w2T, b1, b2, out, NE_);
}
